// SwinTransformer_35450660061706
// MI455X (gfx1250) — matrix operand order repaired and checked by reading, not compiled, not run on hardware
//
#include <hip/hip_runtime.h>
#include <hip/hip_bf16.h>
#include <math.h>

// ---------------- problem constants ----------------
constexpr int CB  = 32;
constexpr int CH  = 56, CW = 56;
constexpr int CC  = 128;
constexpr int CNH = 4;
constexpr int CHD = 32;
constexpr int CWS = 7;
constexpr int CN  = 49;
constexpr int CNP = 64;
constexpr int CNWIN = 64;
constexpr int CTOKENS = CB * CH * CW;       // 100352
constexpr int CTOTW = CB * CNWIN;           // 2048
constexpr int CMPAD = CTOTW * CNP;          // 131072
constexpr float CSCALE = 0.17677669529663687f;

typedef __attribute__((ext_vector_type(16))) _Float16 v16h;
typedef __attribute__((ext_vector_type(8)))  _Float16 v8h;
typedef __attribute__((ext_vector_type(8)))  float    v8f;

// ---------------- CDNA5 async global->LDS copy (CDNA5 ISA) --------
__device__ inline void async_copy_b128(const void* gsrc, void* ldst) {
  unsigned loff = (unsigned)(size_t)ldst;   // low 32 bits of flat ptr = LDS offset
  asm volatile("global_load_async_to_lds_b128 %0, %1, off"
               :: "v"(loff), "v"(gsrc) : "memory");
}
__device__ inline void wait_async() {
  asm volatile("s_wait_asynccnt 0" ::: "memory");
}

// ---------------- WMMA fragment loaders (16-bit layouts, CDNA5 ISA) -------
// A 16x32 (MxK): lanes 0-15 row=lane hold K 0..15;
//                lanes 16-31 row=lane-16 hold K 16..31.
__device__ inline v16h load_a_frag(const _Float16* base, int ld, int row0, int k0, int lane) {
  int row = row0 + (lane & 15);
  int kb  = (lane < 16) ? 0 : 16;
  const _Float16* p = base + row * ld + k0;
  v16h a = {};
#pragma unroll
  for (int v = 0; v < 4; ++v) { a[2*v] = p[kb + 2*v]; a[2*v+1] = p[kb + 2*v + 1]; }
#pragma unroll
  for (int v = 4; v < 8; ++v) { int k = kb + 2*v; a[2*v] = p[k]; a[2*v+1] = p[k+1]; }
  return a;
}
// B 32x16 (KxN) stored N-major: element (k,n) at base[n*ld+k] -> contiguous 16 halves/lane
__device__ inline v16h load_b_frag_nk(const _Float16* base, int ld, int n0, int k0, int lane) {
  int n  = n0 + (lane & 15);
  int kb = k0 + ((lane < 16) ? 0 : 16);
  const _Float16* p = base + n * ld + kb;
  v16h b = {};
#pragma unroll
  for (int v = 0; v < 16; ++v) b[v] = p[v];
  return b;
}

// ---------------- kernels ----------------

__global__ void f32_to_f16_kernel(const float* __restrict__ in, _Float16* __restrict__ out, int n) {
  int i = blockIdx.x * 256 + threadIdx.x;
  if (i < n) out[i] = (_Float16)in[i];
}

__global__ void zero_pad_kernel(_Float16* __restrict__ Xw) {
  int idx = blockIdx.x * 256 + threadIdx.x;              // over 2048*15*128
  int win = idx / (15 * CC);
  int rem = idx % (15 * CC);
  int tok = CN + rem / CC;
  int c   = rem % CC;
  Xw[((size_t)win * CNP + tok) * CC + c] = (_Float16)0.f;
}

// LayerNorm + optional shifted-window pack. mode 0: windowed, mode 1: flat.
__global__ void ln_pack_kernel(const float* __restrict__ xin, const float* __restrict__ g,
                               const float* __restrict__ bt, _Float16* __restrict__ out,
                               int shift, int mode) {
  int wv = threadIdx.x >> 5, lane = threadIdx.x & 31;
  int t = blockIdx.x * 4 + wv;
  if (t >= CTOKENS) return;
  const float* xp = xin + (size_t)t * CC + lane * 4;
  float4 v = *(const float4*)xp;
  float sum = v.x + v.y + v.z + v.w;
  float sq  = v.x*v.x + v.y*v.y + v.z*v.z + v.w*v.w;
#pragma unroll
  for (int o = 16; o > 0; o >>= 1) { sum += __shfl_xor(sum, o); sq += __shfl_xor(sq, o); }
  float mu  = sum * (1.f / CC);
  float var = sq * (1.f / CC) - mu * mu;
  float inv = rsqrtf(var + 1e-5f);
  size_t drow;
  if (mode == 0) {
    int b = t / (CH * CW), s = t % (CH * CW);
    int h = s / CW, w = s % CW;
    int hh = h - shift; if (hh < 0) hh += CH;
    int ww = w - shift; if (ww < 0) ww += CW;
    int win = b * CNWIN + (hh / CWS) * 8 + (ww / CWS);
    int tok = (hh % CWS) * CWS + (ww % CWS);
    drow = (size_t)win * CNP + tok;
  } else {
    drow = (size_t)t;
  }
  _Float16* op = out + drow * CC + lane * 4;
  int c0 = lane * 4;
  op[0] = (_Float16)((v.x - mu) * inv * g[c0+0] + bt[c0+0]);
  op[1] = (_Float16)((v.y - mu) * inv * g[c0+1] + bt[c0+1]);
  op[2] = (_Float16)((v.z - mu) * inv * g[c0+2] + bt[c0+2]);
  op[3] = (_Float16)((v.w - mu) * inv * g[c0+3] + bt[c0+3]);
}

// WMMA GEMM: D = A[MxK] @ W[KxN] + bias. WG tile 64x64, 8 waves (4M x 2N),
// each wave keeps two 16x16 accumulators sharing one A fragment.
// A tile staged via async global->LDS; W tile staged transposed (N-major).
// mode 0: f16 store; 1: f16 GELU; 2: window-reverse + residual -> f32; 3: flat residual -> f32.
__global__ void wmma_gemm_kernel(const _Float16* __restrict__ A, const _Float16* __restrict__ W,
                                 const float* __restrict__ bias, int M, int N, int K,
                                 int mode, const float* __restrict__ resid,
                                 float* __restrict__ outf, _Float16* __restrict__ outh, int shift) {
  __shared__ _Float16 As[64][40];
  __shared__ _Float16 Wt[64][40];   // transposed: Wt[n][k]
  int tid = threadIdx.x;
  int lane = tid & 31, wv = tid >> 5;
  int mi = wv & 3, ni = wv >> 2;
  int m0 = blockIdx.y * 64, n0 = blockIdx.x * 64;
  int arow = tid >> 2, acb = (tid & 3) * 8;      // A: 64 rows x 32 halves, 16B/thread
  int wk = tid >> 3, wn = (tid & 7) * 8;         // W: 32 k-rows x 64 n, 8 halves/thread
  v8f acc0 = {}, acc1 = {};
  for (int k0 = 0; k0 < K; k0 += 32) {
    async_copy_b128(&A[(size_t)(m0 + arow) * K + k0 + acb], &As[arow][acb]);
    v8h wrow = *(const v8h*)&W[(size_t)(k0 + wk) * N + n0 + wn];
#pragma unroll
    for (int j = 0; j < 8; ++j) Wt[wn + j][wk] = wrow[j];
    wait_async();
    __syncthreads();
    v16h a  = load_a_frag(&As[0][0], 40, mi * 16, 0, lane);
    v16h b0 = load_b_frag_nk(&Wt[0][0], 40, ni * 32, 0, lane);
    v16h b1 = load_b_frag_nk(&Wt[0][0], 40, ni * 32 + 16, 0, lane);
    acc0 = __builtin_amdgcn_wmma_f32_16x16x32_f16(false, a, false, b0, (short)0, acc0, false, false);
    acc1 = __builtin_amdgcn_wmma_f32_16x16x32_f16(false, a, false, b1, (short)0, acc1, false, false);
    __syncthreads();
  }
  int colL = lane & 15, rbase = (lane < 16) ? 0 : 8;
  int cg0 = n0 + ni * 32 + colL, cg1 = cg0 + 16;
  float bv0 = bias[cg0], bv1 = bias[cg1];
#pragma unroll
  for (int r = 0; r < 8; ++r) {
    int rg = m0 + mi * 16 + rbase + r;
    float v0 = acc0[r] + bv0;
    float v1 = acc1[r] + bv1;
    if (mode == 0) {
      outh[(size_t)rg * N + cg0] = (_Float16)v0;
      outh[(size_t)rg * N + cg1] = (_Float16)v1;
    } else if (mode == 1) {
      outh[(size_t)rg * N + cg0] = (_Float16)(0.5f * v0 * (1.0f + erff(v0 * 0.70710678f)));
      outh[(size_t)rg * N + cg1] = (_Float16)(0.5f * v1 * (1.0f + erff(v1 * 0.70710678f)));
    } else if (mode == 2) {
      int win = rg >> 6, tok = rg & 63;
      if (tok < CN) {
        int b = win >> 6, wp = win & 63;
        int hh = (wp >> 3) * CWS + tok / CWS;
        int ww = (wp & 7) * CWS + tok % CWS;
        int h = hh + shift; if (h >= CH) h -= CH;
        int w = ww + shift; if (w >= CW) w -= CW;
        size_t base = ((size_t)b * (CH * CW) + h * CW + w) * CC;
        outf[base + cg0] = resid[base + cg0] + v0;
        outf[base + cg1] = resid[base + cg1] + v1;
      }
    } else {
      size_t base = (size_t)rg * N;
      outf[base + cg0] = resid[base + cg0] + v0;
      outf[base + cg1] = resid[base + cg1] + v1;
    }
  }
}

// windowed attention: one workgroup per (window, head), 4 waves
__global__ void attn_kernel(const _Float16* __restrict__ QKV, const float* __restrict__ rpb,
                            _Float16* __restrict__ AO, int shift) {
  __shared__ _Float16 qs[CNP][CHD];
  __shared__ _Float16 ks[CNP][CHD];
  __shared__ _Float16 vsT[CHD][72];   // transposed V: vsT[d][j]
  __shared__ float    Sd[CNP][CNP];
  __shared__ _Float16 Ps[CNP][CNP];
  int win = blockIdx.x, head = blockIdx.y;
  int tid = threadIdx.x, lane = tid & 31, wv = tid >> 5;
  size_t rowbase = (size_t)win * CNP;

  // q, k: async 16-byte chunks at half offsets 0 and 16 (64 rows x 2 chunks x 2 mats = 256 transfers)
  for (int id = tid; id < 256; id += 128) {
    int mat = id >> 7;                 // 0 = q, 1 = k
    int rem = id & 127;
    int r = rem >> 1, hc = (rem & 1) * 16;
    const _Float16* src = QKV + (rowbase + r) * (3 * CC) + mat * CC + head * CHD + hc;
    _Float16* dst = mat ? &ks[r][hc] : &qs[r][hc];
    async_copy_b128(src, dst);
  }
  // v: scalar transposed store
  for (int idx = tid; idx < CNP * CHD; idx += 128) {
    int r = idx >> 5, c = idx & 31;
    vsT[c][r] = QKV[(rowbase + r) * (3 * CC) + 2 * CC + head * CHD + c];
  }
  wait_async();
  __syncthreads();

  int wp = win & 63, wh = wp >> 3, wwv = wp & 7;
  int mi = wv;
  int colL = lane & 15, rbase = (lane < 16) ? 0 : 8;

  // S = (q @ k^T) * scale + rel-pos-bias (+ shift mask)
  for (int ni = 0; ni < 4; ++ni) {
    v16h a  = load_a_frag(&qs[0][0], CHD, mi * 16, 0, lane);
    v16h bf = load_b_frag_nk(&ks[0][0], CHD, ni * 16, 0, lane);  // (d,j) at ks[j][d]
    v8f c = {};
    c = __builtin_amdgcn_wmma_f32_16x16x32_f16(false, a, false, bf, (short)0, c, false, false);
#pragma unroll
    for (int r = 0; r < 8; ++r) {
      int i = mi * 16 + rbase + r, j = ni * 16 + colL;
      float val;
      if (j >= CN) val = -1e30f;
      else if (i >= CN) val = 0.f;
      else {
        int ri = i / CWS, ci = i % CWS, rj = j / CWS, cj = j % CWS;
        int ridx = (ri - rj + CWS - 1) * (2 * CWS - 1) + (ci - cj + CWS - 1);
        val = c[r] * CSCALE + rpb[ridx * CNH + head];
        if (shift) {
          int hi = wh * CWS + ri, wi = wwv * CWS + ci;
          int hj = wh * CWS + rj, wj = wwv * CWS + cj;
          int regi = (hi < 49 ? 0 : (hi < 53 ? 1 : 2)) * 3 + (wi < 49 ? 0 : (wi < 53 ? 1 : 2));
          int regj = (hj < 49 ? 0 : (hj < 53 ? 1 : 2)) * 3 + (wj < 49 ? 0 : (wj < 53 ? 1 : 2));
          if (regi != regj) val -= 100.f;
        }
      }
      Sd[i][j] = val;
    }
  }
  __syncthreads();

  // row softmax
  if (tid < CNP) {
    float mx = -1e30f;
    for (int j = 0; j < CNP; ++j) mx = fmaxf(mx, Sd[tid][j]);
    float sum = 0.f;
    for (int j = 0; j < CNP; ++j) { float e = __expf(Sd[tid][j] - mx); Sd[tid][j] = e; sum += e; }
    float inv = 1.f / sum;
    for (int j = 0; j < CNP; ++j) Ps[tid][j] = (_Float16)(Sd[tid][j] * inv);
  }
  __syncthreads();

  // O = P @ V
  for (int ni = 0; ni < 2; ++ni) {
    v8f c = {};
#pragma unroll
    for (int kk = 0; kk < 2; ++kk) {
      v16h a  = load_a_frag(&Ps[0][0], CNP, mi * 16, kk * 32, lane);
      v16h bf = load_b_frag_nk(&vsT[0][0], 72, ni * 16, kk * 32, lane); // (j,d) at vsT[d][j]
      c = __builtin_amdgcn_wmma_f32_16x16x32_f16(false, a, false, bf, (short)0, c, false, false);
    }
#pragma unroll
    for (int r = 0; r < 8; ++r) {
      int i = mi * 16 + rbase + r;
      int d = ni * 16 + colL;
      AO[(rowbase + i) * CC + head * CHD + d] = (_Float16)c[r];
    }
  }
}

// ---------------- host launcher ----------------
extern "C" void kernel_launch(void* const* d_in, const int* in_sizes, int n_in,
                              void* d_out, int out_size, void* d_ws, size_t ws_size,
                              hipStream_t stream) {
  const float* x_in  = (const float*)d_in[0];
  const float* n1g   = (const float*)d_in[1];
  const float* n1b   = (const float*)d_in[2];
  const float* qkvw  = (const float*)d_in[3];
  const float* qkvb  = (const float*)d_in[4];
  const float* rpb   = (const float*)d_in[5];
  const float* pw    = (const float*)d_in[6];
  const float* pb    = (const float*)d_in[7];
  const float* n2g   = (const float*)d_in[8];
  const float* n2b   = (const float*)d_in[9];
  const float* f1w   = (const float*)d_in[10];
  const float* f1b   = (const float*)d_in[11];
  const float* f2w   = (const float*)d_in[12];
  const float* f2b   = (const float*)d_in[13];

  char* ws = (char*)d_ws;
  size_t off = 0;
  auto take = [&](size_t bytes) -> void* {
    void* p = ws + off;
    off = (off + bytes + 255) & ~(size_t)255;
    return p;
  };

  float*    bufx0 = (float*)   take((size_t)CTOKENS * CC * 4);
  float*    xout  = (float*)   take((size_t)CTOKENS * CC * 4);
  _Float16* Xw    = (_Float16*)take((size_t)CMPAD * CC * 2);
  _Float16* QKV   = (_Float16*)take((size_t)CMPAD * 3 * CC * 2);
  _Float16* AO    = (_Float16*)take((size_t)CMPAD * CC * 2);
  _Float16* Hw    = (_Float16*)take((size_t)CTOKENS * CC * 2);
  _Float16* Hid   = (_Float16*)take((size_t)CTOKENS * 4 * CC * 2);
  _Float16* Wq16  = (_Float16*)take((size_t)CC * 3 * CC * 2);
  _Float16* Wp16  = (_Float16*)take((size_t)CC * CC * 2);
  _Float16* W116  = (_Float16*)take((size_t)CC * 4 * CC * 2);
  _Float16* W216  = (_Float16*)take((size_t)4 * CC * CC * 2);

  for (int i = 0; i < 2; ++i) {
    int shift = i ? (CWS / 2) : 0;
    const float* xin = i ? bufx0 : x_in;
    float* dst = i ? (float*)d_out : bufx0;

    const float* qkvw_i = qkvw + (size_t)i * CC * 3 * CC;
    const float* qkvb_i = qkvb + (size_t)i * 3 * CC;
    const float* rpb_i  = rpb  + (size_t)i * (2*CWS-1)*(2*CWS-1) * CNH;
    const float* pw_i   = pw   + (size_t)i * CC * CC;
    const float* pb_i   = pb   + (size_t)i * CC;
    const float* f1w_i  = f1w  + (size_t)i * CC * 4 * CC;
    const float* f1b_i  = f1b  + (size_t)i * 4 * CC;
    const float* f2w_i  = f2w  + (size_t)i * 4 * CC * CC;
    const float* f2b_i  = f2b  + (size_t)i * CC;

    f32_to_f16_kernel<<<(CC*3*CC + 255)/256, 256, 0, stream>>>(qkvw_i, Wq16, CC*3*CC);
    f32_to_f16_kernel<<<(CC*CC   + 255)/256, 256, 0, stream>>>(pw_i,   Wp16, CC*CC);
    f32_to_f16_kernel<<<(CC*4*CC + 255)/256, 256, 0, stream>>>(f1w_i,  W116, CC*4*CC);
    f32_to_f16_kernel<<<(4*CC*CC + 255)/256, 256, 0, stream>>>(f2w_i,  W216, 4*CC*CC);

    zero_pad_kernel<<<(CTOTW*15*CC)/256, 256, 0, stream>>>(Xw);
    ln_pack_kernel<<<CTOKENS/4, 128, 0, stream>>>(xin, n1g + (size_t)i*CC, n1b + (size_t)i*CC,
                                                  Xw, shift, 0);

    // QKV projection: [131072 x 128] @ [128 x 384]
    wmma_gemm_kernel<<<dim3(3*CC/64, CMPAD/64), 256, 0, stream>>>(
        Xw, Wq16, qkvb_i, CMPAD, 3*CC, CC, 0, nullptr, nullptr, QKV, 0);

    attn_kernel<<<dim3(CTOTW, CNH), 128, 0, stream>>>(QKV, rpb_i, AO, shift);

    // output projection + window-reverse + residual
    wmma_gemm_kernel<<<dim3(CC/64, CMPAD/64), 256, 0, stream>>>(
        AO, Wp16, pb_i, CMPAD, CC, CC, 2, xin, xout, nullptr, shift);

    ln_pack_kernel<<<CTOKENS/4, 128, 0, stream>>>(xout, n2g + (size_t)i*CC, n2b + (size_t)i*CC,
                                                  Hw, 0, 1);

    // MLP fc1 + GELU
    wmma_gemm_kernel<<<dim3(4*CC/64, CTOKENS/64), 256, 0, stream>>>(
        Hw, W116, f1b_i, CTOKENS, 4*CC, CC, 1, nullptr, nullptr, Hid, 0);

    // MLP fc2 + residual
    wmma_gemm_kernel<<<dim3(CC/64, CTOKENS/64), 256, 0, stream>>>(
        Hid, W216, f2b_i, CTOKENS, CC, 4*CC, 3, xout, dst, nullptr, 0);
  }
  (void)in_sizes; (void)n_in; (void)out_size; (void)ws_size;
}
